// DifferenceEncodingGRU_13168369729786
// MI455X (gfx1250) — matrix operand order repaired and checked by reading, not compiled, not run on hardware
//
#include <hip/hip_runtime.h>
#include <math.h>

// Problem sizes (match reference)
constexpr int Hh = 1024;   // hidden
constexpr int In = 64;     // input
constexpr int Ou = 64;     // output
constexpr int Au = 192;    // augmented input
constexpr int Bb = 128;    // batch
constexpr int Ss = 512;    // seq len

// Decomposition
constexpr int NG  = 16;        // persistent workgroups (hidden dim split)
constexpr int HWs = Hh / NG;   // 64 hidden units per WG
constexpr int NT  = 512;       // 16 wave32s per WG

typedef __attribute__((ext_vector_type(16))) __bf16 bf16x16;
typedef __attribute__((ext_vector_type(8)))  float  f32x8;

union FragB16 { bf16x16 v; uint4 q[2]; __bf16 e[16]; };

__device__ __forceinline__ f32x8 wmma_bf16(const FragB16& a, const FragB16& b, f32x8 c) {
  // D = A(16x32 bf16) * B(32x16 bf16) + C(16x16 f32)
  return __builtin_amdgcn_wmma_f32_16x16x32_bf16(false, a.v, false, b.v, (short)0, c, false, false);
}

// Workspace layout (needs ~9.2 MB)
constexpr size_t OFF_CTR  = 0;                                          // 512 x u32 barrier counters
constexpr size_t OFF_WIH  = 4096;                                       // [3H][A]  bf16
constexpr size_t OFF_WHH  = OFF_WIH  + (size_t)3*Hh*Au*sizeof(__bf16);  // [3H][H]  bf16
constexpr size_t OFF_WOUT = OFF_WHH  + (size_t)3*Hh*Hh*sizeof(__bf16);  // [O][H]   bf16
constexpr size_t OFF_HBUF = OFF_WOUT + (size_t)Ou*Hh*sizeof(__bf16);    // [2][B][H] bf16 (double buffered h)
constexpr size_t OFF_PART = OFF_HBUF + (size_t)2*Bb*Hh*sizeof(__bf16);  // [2][NG][B][O] f32 partials of h*W_out^T

// ---------------------------------------------------------------------------
// Init: zero barrier counters, convert weights fp32 -> bf16 (row-major kept:
// B-fragment of W^T == contiguous K-run of a row-major W row).
// ---------------------------------------------------------------------------
__global__ void gru_init(const float* __restrict__ Wih, const float* __restrict__ Whh,
                         const float* __restrict__ Wout, unsigned char* __restrict__ ws) {
  size_t idx = (size_t)blockIdx.x * blockDim.x + threadIdx.x;
  unsigned* ctr  = (unsigned*)(ws + OFF_CTR);
  __bf16*   wih  = (__bf16*)(ws + OFF_WIH);
  __bf16*   whh  = (__bf16*)(ws + OFF_WHH);
  __bf16*   wout = (__bf16*)(ws + OFF_WOUT);
  if (idx < (size_t)Ss)        ctr[idx]  = 0u;
  if (idx < (size_t)3*Hh*Au)   wih[idx]  = (__bf16)Wih[idx];
  if (idx < (size_t)3*Hh*Hh)   whh[idx]  = (__bf16)Whh[idx];
  if (idx < (size_t)Ou*Hh)     wout[idx] = (__bf16)Wout[idx];
}

// ---------------------------------------------------------------------------
// Persistent GRU kernel: 16 WGs, one device-wide barrier per time step.
// WG g owns hidden units [g*64, g*64+64).
// Gate GEMMs: wave w -> j-tile = w>>2, m-tiles {2*(w&3), 2*(w&3)+1}.
// Out GEMM:   wave w -> m-tile = w&7, o-tiles {2*(w>>3), 2*(w>>3)+1}.
// ---------------------------------------------------------------------------
__global__ __launch_bounds__(NT)
void gru_persistent(const float* __restrict__ X0, const float* __restrict__ Vin,
                    const float* __restrict__ b_ih, const float* __restrict__ b_hh,
                    const float* __restrict__ b_out, float* __restrict__ out,
                    unsigned char* __restrict__ ws)
{
  const int g    = blockIdx.x;
  const int tid  = threadIdx.x;
  const int w    = tid >> 5;
  const int lane = tid & 31;
  const int ln16 = lane & 15;
  const int hi   = lane >> 4;

  unsigned*     ctr  = (unsigned*)(ws + OFF_CTR);
  const __bf16* Wih  = (const __bf16*)(ws + OFF_WIH);
  const __bf16* Whh  = (const __bf16*)(ws + OFF_WHH);
  const __bf16* Wo   = (const __bf16*)(ws + OFF_WOUT);
  __bf16*       hbuf = (__bf16*)(ws + OFF_HBUF);
  float*        part = (float*)(ws + OFF_PART);

  __shared__ float  x_lds[Bb * Ou];      // x_{t-1}, fp32 (32 KB)
  __shared__ __bf16 hnew_lds[Bb * HWs];  // this step's h slice, bf16 (16 KB)

  // gate-GEMM mapping
  const int jt   = w >> 2;
  const int mp   = w & 3;
  const int jcol = jt * 16 + ln16;       // [0,64) within slice; this lane's N coord
  const int jabs = g * HWs + jcol;       // [0,1024)

  const float bR  = b_ih[jabs] + b_hh[jabs];
  const float bZ  = b_ih[Hh + jabs] + b_hh[Hh + jabs];
  const float bIN = b_ih[2*Hh + jabs];
  const float bHN = b_hh[2*Hh + jabs];

  // out-GEMM mapping
  const int mt2 = w & 7;
  const int otp = w >> 3;

  // fp32 h_prev slice lives in registers in C/D fragment layout across steps
  f32x8 hp[2];
  #pragma unroll
  for (int i = 0; i < 2; ++i)
    #pragma unroll
    for (int v = 0; v < 8; ++v) hp[i][v] = 0.f;

  const __bf16* WihR = Wih + (size_t)jabs * Au;
  const __bf16* WihZ = Wih + (size_t)(Hh + jabs) * Au;
  const __bf16* WihN = Wih + (size_t)(2*Hh + jabs) * Au;
  const __bf16* WhhR = Whh + (size_t)jabs * Hh;
  const __bf16* WhhZ = Whh + (size_t)(Hh + jabs) * Hh;
  const __bf16* WhhN = Whh + (size_t)(2*Hh + jabs) * Hh;

  for (int t = 0; t <= Ss; ++t) {
    // ---------------- Phase A: finalize x_{t-1}, write output ----------------
    {
      const int m  = tid >> 2;
      const int o0 = (tid & 3) << 4;
      float* xl = x_lds + m * Ou + o0;
      if (t == 0) {
        const float* p = X0 + m * Ou + o0;
        #pragma unroll
        for (int i = 0; i < 16; ++i) xl[i] = p[i];
      } else {
        // x_{t-1} = x_{t-2} + v_{t-1} + sum_g partial_g + b_out
        const float* vrow = Vin + (size_t)m * Ss * In + (size_t)(t - 1) * In + o0;
        const float* bo   = b_out + o0;
        float acc[16];
        #pragma unroll
        for (int i = 0; i < 16; ++i) acc[i] = xl[i] + vrow[i] + bo[i];
        const float* pprev = part + (size_t)((t - 1) & 1) * NG * Bb * Ou + m * Ou + o0;
        for (int gg = 0; gg < NG; ++gg) {
          const float4* pp = (const float4*)(pprev + (size_t)gg * Bb * Ou);
          #pragma unroll
          for (int q = 0; q < 4; ++q) {
            float4 pv = pp[q];
            acc[q*4+0] += pv.x; acc[q*4+1] += pv.y; acc[q*4+2] += pv.z; acc[q*4+3] += pv.w;
          }
        }
        float* od = out + (size_t)m * Ss * Ou + (size_t)(t - 1) * Ou + o0;
        #pragma unroll
        for (int i = 0; i < 16; ++i) { od[i] = acc[i]; xl[i] = acc[i]; }
      }
    }
    __syncthreads();
    if (t == Ss) break;

    // ---------------- Phase B: gate GEMMs ----------------
    f32x8 accR[2], accZ[2], accIN[2], accHN[2];
    #pragma unroll
    for (int i = 0; i < 2; ++i)
      #pragma unroll
      for (int v = 0; v < 8; ++v) { accR[i][v]=0.f; accZ[i][v]=0.f; accIN[i][v]=0.f; accHN[i][v]=0.f; }

    // gi: aug[128,192] x W_ih_slice^T   (aug built on the fly: [v_t | x | x+v_t])
    #pragma unroll
    for (int kk = 0; kk < Au; kk += 32) {
      FragB16 fa[2], fbR, fbZ, fbN;
      const int colB = kk + hi * 8;
      fbR.q[0] = *(const uint4*)(WihR + colB); fbR.q[1] = *(const uint4*)(WihR + colB + 16);
      fbZ.q[0] = *(const uint4*)(WihZ + colB); fbZ.q[1] = *(const uint4*)(WihZ + colB + 16);
      fbN.q[0] = *(const uint4*)(WihN + colB); fbN.q[1] = *(const uint4*)(WihN + colB + 16);
      #pragma unroll
      for (int mi = 0; mi < 2; ++mi) {
        const int row = (mp * 2 + mi) * 16 + ln16;
        const float* vrow = Vin + (size_t)row * Ss * In + (size_t)t * In;
        const float* xrow = x_lds + row * Ou;
        #pragma unroll
        for (int hf = 0; hf < 2; ++hf) {
          const int kb  = kk + hf * 16 + hi * 8;  // 8-aligned, never crosses a segment
          const int seg = kb >> 6;
          const int off = kb & 63;
          float va[8];
          if (seg == 0) {
            #pragma unroll
            for (int e2 = 0; e2 < 8; ++e2) va[e2] = vrow[off + e2];
          } else if (seg == 1) {
            #pragma unroll
            for (int e2 = 0; e2 < 8; ++e2) va[e2] = xrow[off + e2];
          } else {
            #pragma unroll
            for (int e2 = 0; e2 < 8; ++e2) va[e2] = xrow[off + e2] + vrow[off + e2];
          }
          #pragma unroll
          for (int e2 = 0; e2 < 8; ++e2) fa[mi].e[hf * 8 + e2] = (__bf16)va[e2];
        }
      }
      #pragma unroll
      for (int mi = 0; mi < 2; ++mi) {
        accR[mi]  = wmma_bf16(fa[mi], fbR, accR[mi]);
        accZ[mi]  = wmma_bf16(fa[mi], fbZ, accZ[mi]);
        accIN[mi] = wmma_bf16(fa[mi], fbN, accIN[mi]);
      }
    }

    // gh: h_{t-1}[128,1024] x W_hh_slice^T   (h_{-1} == 0 -> skipped at t==0)
    if (t > 0) {
      const __bf16* hprev_g = hbuf + (size_t)((t - 1) & 1) * Bb * Hh;
      for (int kk = 0; kk < Hh; kk += 32) {
        FragB16 fa[2], fbR, fbZ, fbN;
        const int colB = kk + hi * 8;
        fbR.q[0] = *(const uint4*)(WhhR + colB); fbR.q[1] = *(const uint4*)(WhhR + colB + 16);
        fbZ.q[0] = *(const uint4*)(WhhZ + colB); fbZ.q[1] = *(const uint4*)(WhhZ + colB + 16);
        fbN.q[0] = *(const uint4*)(WhhN + colB); fbN.q[1] = *(const uint4*)(WhhN + colB + 16);
        #pragma unroll
        for (int mi = 0; mi < 2; ++mi) {
          const int row = (mp * 2 + mi) * 16 + ln16;
          const __bf16* hrow = hprev_g + (size_t)row * Hh + kk + hi * 8;
          fa[mi].q[0] = *(const uint4*)(hrow);
          fa[mi].q[1] = *(const uint4*)(hrow + 16);
        }
        #pragma unroll
        for (int mi = 0; mi < 2; ++mi) {
          accR[mi]  = wmma_bf16(fa[mi], fbR, accR[mi]);
          accZ[mi]  = wmma_bf16(fa[mi], fbZ, accZ[mi]);
          accHN[mi] = wmma_bf16(fa[mi], fbN, accHN[mi]);
        }
      }
    }

    // ---------------- Gates + h update (fragment layout) ----------------
    __bf16* hcur_g = hbuf + (size_t)(t & 1) * Bb * Hh;
    #pragma unroll
    for (int mi = 0; mi < 2; ++mi) {
      const int mbase = (mp * 2 + mi) * 16 + hi * 8;
      #pragma unroll
      for (int v = 0; v < 8; ++v) {
        const float r  = 1.f / (1.f + __expf(-(accR[mi][v] + bR)));
        const float z  = 1.f / (1.f + __expf(-(accZ[mi][v] + bZ)));
        const float nn = tanhf(accIN[mi][v] + bIN + r * (accHN[mi][v] + bHN));
        const float hv = (1.f - z) * nn + z * hp[mi][v];
        hp[mi][v] = hv;                       // fp32 h_prev stays in registers
        const __bf16 hb = (__bf16)hv;
        const int m = mbase + v;
        hcur_g[(size_t)m * Hh + g * HWs + jcol] = hb;   // publish bf16 h slice
        hnew_lds[m * HWs + jcol] = hb;                  // stage for out-GEMM
      }
    }
    __syncthreads();

    // ---------------- Phase C: partial x_t = h_slice x W_out_slice^T ----------------
    f32x8 po[2];
    #pragma unroll
    for (int q = 0; q < 2; ++q)
      #pragma unroll
      for (int v = 0; v < 8; ++v) po[q][v] = 0.f;

    const __bf16* WoA = Wo + (size_t)(otp * 32 + ln16)      * Hh + g * HWs;
    const __bf16* WoB = Wo + (size_t)(otp * 32 + 16 + ln16) * Hh + g * HWs;
    #pragma unroll
    for (int kk = 0; kk < HWs; kk += 32) {
      FragB16 fa, fb0, fb1;
      const __bf16* hrow = hnew_lds + (mt2 * 16 + ln16) * HWs + kk + hi * 8;
      fa.q[0] = *(const uint4*)(hrow);
      fa.q[1] = *(const uint4*)(hrow + 16);
      const int colB = kk + hi * 8;
      fb0.q[0] = *(const uint4*)(WoA + colB); fb0.q[1] = *(const uint4*)(WoA + colB + 16);
      fb1.q[0] = *(const uint4*)(WoB + colB); fb1.q[1] = *(const uint4*)(WoB + colB + 16);
      po[0] = wmma_bf16(fa, fb0, po[0]);
      po[1] = wmma_bf16(fa, fb1, po[1]);
    }
    float* pcur = part + (size_t)(t & 1) * NG * Bb * Ou + (size_t)g * Bb * Ou;
    #pragma unroll
    for (int q = 0; q < 2; ++q) {
      const int o = (otp * 2 + q) * 16 + ln16;
      #pragma unroll
      for (int v = 0; v < 8; ++v) {
        const int m = mt2 * 16 + hi * 8 + v;
        pcur[m * Ou + o] = po[q][v];
      }
    }

    // ---------------- Phase D: device-wide barrier for step t ----------------
    __threadfence();     // release h slice + partials
    __syncthreads();
    if (tid == 0) {
      __hip_atomic_fetch_add(&ctr[t], 1u, __ATOMIC_RELEASE, __HIP_MEMORY_SCOPE_AGENT);
      while (__hip_atomic_load(&ctr[t], __ATOMIC_ACQUIRE, __HIP_MEMORY_SCOPE_AGENT) < (unsigned)NG)
        __builtin_amdgcn_s_sleep(1);
    }
    __syncthreads();
    __threadfence();     // acquire other WGs' h slices + partials
  }
}

// ---------------------------------------------------------------------------
extern "C" void kernel_launch(void* const* d_in, const int* in_sizes, int n_in,
                              void* d_out, int out_size, void* d_ws, size_t ws_size,
                              hipStream_t stream) {
  const float* X0   = (const float*)d_in[0];
  const float* V    = (const float*)d_in[1];
  const float* Wih  = (const float*)d_in[2];
  const float* Whh  = (const float*)d_in[3];
  const float* bih  = (const float*)d_in[4];
  const float* bhh  = (const float*)d_in[5];
  const float* Wout = (const float*)d_in[6];
  const float* bout = (const float*)d_in[7];
  float* out = (float*)d_out;
  unsigned char* ws = (unsigned char*)d_ws;

  // Re-run init every call: harness poisons d_ws before timed replays.
  const int nconv = 3 * Hh * Hh;  // largest converted array
  gru_init<<<(nconv + 255) / 256, 256, 0, stream>>>(Wih, Whh, Wout, ws);

  gru_persistent<<<dim3(NG), dim3(NT), 0, stream>>>(X0, V, bih, bhh, bout, out, ws);
}
